// DynChComb_57569741635806
// MI455X (gfx1250) — matrix operand order repaired and checked by reading, not compiled, not run on hardware
//
#include <hip/hip_runtime.h>

// DynChComb on MI455X (gfx1250, wave32):
//   pool -> tiny MLP -> softmax/T -> mix conv_w into per-sample Wb (bf16)
//   -> batched GEMM via v_wmma_f32_16x16x32_bf16 fused with LayerNorm(S) + erf-GELU.
// conv_b is provably a no-op: a per-(b,o) constant shift along S cancels in LayerNorm.

typedef __attribute__((ext_vector_type(16))) __bf16 bf16x16;
typedef __attribute__((ext_vector_type(2)))  __bf16 bf16x2;
typedef __attribute__((ext_vector_type(8)))  float  f32x8;
typedef __attribute__((ext_vector_type(2)))  float  f32x2;

#define TID ((int)threadIdx.x)

// Pair conversion expressed as a vector convert -> single v_cvt_pk_bf16_f32 vD, vLo, vHi.
__device__ __forceinline__ unsigned pack_bf2(float lo, float hi) {
    f32x2 f; f.x = lo; f.y = hi;
    bf16x2 h = __builtin_convertvector(f, bf16x2);
    union { bf16x2 h; unsigned u; } v; v.h = h;
    return v.u;
}
__device__ __forceinline__ float bf2f(unsigned short h) {
    union { unsigned u; float f; } v; v.u = ((unsigned)h) << 16;
    return v.f;
}

struct FragBF { union { bf16x16 v; uint4 q[2]; }; };

// ---------------- Stage 1: global average pool over S ----------------
__global__ __launch_bounds__(256) void dyn_pool(const float* __restrict__ x,
                                                float* __restrict__ pooled) {
    __shared__ float red[256];
    const int c = blockIdx.x, b = blockIdx.y;
    const float4* row = reinterpret_cast<const float4*>(x + ((size_t)(b * 256 + c)) * 4096);
    float s = 0.f;
    for (int i = TID; i < 1024; i += 256) {
        float4 f = row[i];
        s += (f.x + f.y) + (f.z + f.w);
    }
    red[TID] = s; __syncthreads();
    for (int w = 128; w > 0; w >>= 1) { if (TID < w) red[TID] += red[TID + w]; __syncthreads(); }
    if (TID == 0) pooled[b * 256 + c] = red[0] * (1.0f / 4096.0f);
}

// ---------------- Stage 2: attention head (FC-ReLU-FC-softmax/T) ----------------
__global__ __launch_bounds__(256) void dyn_attn(const float* __restrict__ pooled,
                                                const float* __restrict__ w1,
                                                const float* __restrict__ w2,
                                                const float* __restrict__ b2,
                                                float* __restrict__ att) {
    __shared__ float sp[256]; __shared__ float sh[64]; __shared__ float sl[4];
    const int b = blockIdx.x;
    sp[TID] = pooled[b * 256 + TID];
    __syncthreads();
    if (TID < 64) {
        float acc = 0.f;
        const float* wr = w1 + TID * 256;
        for (int c = 0; c < 256; ++c) acc += sp[c] * wr[c];
        sh[TID] = fmaxf(acc, 0.f);
    }
    __syncthreads();
    if (TID < 4) {
        float acc = b2[TID];
        const float* wr = w2 + TID * 64;
        for (int j = 0; j < 64; ++j) acc += sh[j] * wr[j];
        sl[TID] = acc * (1.0f / 30.0f);   // temperature
    }
    __syncthreads();
    if (TID == 0) {
        float m = fmaxf(fmaxf(sl[0], sl[1]), fmaxf(sl[2], sl[3]));
        float e0 = expf(sl[0] - m), e1 = expf(sl[1] - m), e2 = expf(sl[2] - m), e3 = expf(sl[3] - m);
        float inv = 1.0f / (e0 + e1 + e2 + e3);
        att[b * 4 + 0] = e0 * inv; att[b * 4 + 1] = e1 * inv;
        att[b * 4 + 2] = e2 * inv; att[b * 4 + 3] = e3 * inv;
    }
}

// ---------------- Stage 3: mix expert weights -> per-sample Wb (bf16, pair-packed) ------
__global__ __launch_bounds__(256) void dyn_mix(const float* __restrict__ att,
                                               const float* __restrict__ conv_w,
                                               unsigned* __restrict__ Wb32) {
    const size_t p   = (size_t)blockIdx.x * 256 + TID;   // dword index: B*Cout*Cin/2
    const int b   = (int)(p >> 16);                      // 65536 dwords per sample
    const int rem = (int)(p & 65535);                    // float2 index within sample
    const float a0 = att[b * 4 + 0], a1 = att[b * 4 + 1];
    const float a2 = att[b * 4 + 2], a3 = att[b * 4 + 3];
    const float2* cw = reinterpret_cast<const float2*>(conv_w);
    float2 c0 = cw[rem], c1 = cw[65536 + rem], c2 = cw[131072 + rem], c3 = cw[196608 + rem];
    float w0 = a0 * c0.x + a1 * c1.x + a2 * c2.x + a3 * c3.x;
    float w1 = a0 * c0.y + a1 * c1.y + a2 * c2.y + a3 * c3.y;
    Wb32[p] = pack_bf2(w0, w1);
}

// ---------------- Stage 4: batched GEMM (WMMA bf16) fused with LayerNorm + GELU --------
// Workgroup: one batch b, 32 Cout rows, full S=4096. 256 threads = 8 waves.
// LDS: x-stage [64][264] bf16 (33792 B) + y tile COLUMN-major [4096][34] bf16 (278528 B)
//      + stats (512 B) = 312832 B < 320 KB/WGP.
__global__ __launch_bounds__(256) void dyn_main(const float* __restrict__ x,
                                                const unsigned short* __restrict__ Wb,
                                                const float* __restrict__ gamma,
                                                const float* __restrict__ beta,
                                                float* __restrict__ out) {
    extern __shared__ char smem[];
    unsigned short* xs   = (unsigned short*)smem;                  // [64][264] bf16, s-major
    unsigned*       xs32 = (unsigned*)smem;                        // dword view, pitch 132
    unsigned short* yT   = (unsigned short*)(smem + 33792);        // [4096][34] bf16, col-major
    unsigned*       yT32 = (unsigned*)(smem + 33792);              // dword view, pitch 17
    float* sums = (float*)(smem + 33792 + 278528);                 // [32]
    float* sqrs = sums + 32;
    float* mean = sqrs + 32;
    float* rstd = mean + 32;

    const int b    = blockIdx.y;
    const int o0   = blockIdx.x * 32;
    const int lane = TID & 31, wave = TID >> 5;
    const int half = lane >> 4, lq = lane & 15;
    const int mi   = wave >> 2, ni = wave & 3;     // 2 M-tiles x 4 N-tiles per chunk

    if (TID < 32) { sums[TID] = 0.f; sqrs[TID] = 0.f; }

    // Preload this wave's A fragments (Wb rows o0+mi*16..+15, K=256) once.
    // ISA 16-bit A layout: lane M=lq; half 0 holds K {0-7,16-23}, half 1 holds K {8-15,24-31}.
    FragBF a[8];
    {
        const unsigned short* wrow = Wb + ((size_t)(b * 512 + o0 + mi * 16 + lq)) * 256;
#pragma unroll
        for (int kt = 0; kt < 8; ++kt) {
            a[kt].q[0] = *reinterpret_cast<const uint4*>(wrow + kt * 32 + half * 8);
            a[kt].q[1] = *reinterpret_cast<const uint4*>(wrow + kt * 32 + 16 + half * 8);
        }
    }

    float rowSum[8], rowSqr[8];
#pragma unroll
    for (int r = 0; r < 8; ++r) { rowSum[r] = 0.f; rowSqr[r] = 0.f; }

    const float* xb = x + (size_t)b * (256 * 4096);

    for (int ch = 0; ch < 64; ++ch) {
        const int s0 = ch * 64;
        __syncthreads();                            // xs free from previous chunk's readers
        // Stage x[b][:,s0:s0+64] fp32 -> bf16, transposed to [s_local][c].
        // Each thread: 2 channels x 4 s -> 2 float4 loads, 4 packed b32 LDS stores.
#pragma unroll
        for (int it = 0; it < 8; ++it) {
            int e  = TID + it * 256;                // 2048 (channel-pair, s-quad) units
            int cp = e >> 4;                        // channel pair 0..127
            int sl = (e & 15) * 4;
            const float* p0 = xb + (size_t)(2 * cp) * 4096 + s0 + sl;
            float4 fa = *reinterpret_cast<const float4*>(p0);
            float4 fb = *reinterpret_cast<const float4*>(p0 + 4096);
            xs32[(sl + 0) * 132 + cp] = pack_bf2(fa.x, fb.x);
            xs32[(sl + 1) * 132 + cp] = pack_bf2(fa.y, fb.y);
            xs32[(sl + 2) * 132 + cp] = pack_bf2(fa.z, fb.z);
            xs32[(sl + 3) * 132 + cp] = pack_bf2(fa.w, fb.w);
        }
        __syncthreads();

        f32x8 acc = {0.f, 0.f, 0.f, 0.f, 0.f, 0.f, 0.f, 0.f};
        const unsigned short* pbase = xs + (ni * 16 + lq) * 264 + half * 8;
        // Software-pipelined K loop: load fragment kt+1 while wmma kt executes.
        FragBF bfr[2];
        bfr[0].q[0] = *reinterpret_cast<const uint4*>(pbase);
        bfr[0].q[1] = *reinterpret_cast<const uint4*>(pbase + 16);
#pragma unroll
        for (int kt = 0; kt < 8; ++kt) {
            if (kt < 7) {
                const unsigned short* p = pbase + (kt + 1) * 32;
                bfr[(kt + 1) & 1].q[0] = *reinterpret_cast<const uint4*>(p);
                bfr[(kt + 1) & 1].q[1] = *reinterpret_cast<const uint4*>(p + 16);
            }
            acc = __builtin_amdgcn_wmma_f32_16x16x32_bf16(
                false, a[kt].v, false, bfr[kt & 1].v, (short)0, acc, false, false);
        }

        // Column-major y tile: lane's 8 row-values are contiguous -> 4 packed b32 stores.
        const int sCol = s0 + ni * 16 + lq;
        unsigned* ydst = yT32 + sCol * 17 + mi * 8 + half * 4;
#pragma unroll
        for (int j = 0; j < 4; ++j) {
            float v0 = acc[2 * j], v1 = acc[2 * j + 1];
            rowSum[2 * j]     += v0; rowSqr[2 * j]     += v0 * v0;
            rowSum[2 * j + 1] += v1; rowSqr[2 * j + 1] += v1 * v1;
            ydst[j] = pack_bf2(v0, v1);
        }
    }

    __syncthreads();
#pragma unroll
    for (int r = 0; r < 8; ++r) {                   // ds_add_f32 reductions
        atomicAdd(&sums[mi * 16 + half * 8 + r], rowSum[r]);
        atomicAdd(&sqrs[mi * 16 + half * 8 + r], rowSqr[r]);
    }
    // Reuse xs region (33.8 KB) to stage gamma/beta (32 KB), vectorized.
    float* gS = (float*)smem;
    float* bS = gS + 4096;
    for (int s4 = TID * 4; s4 < 4096; s4 += 1024) {
        *reinterpret_cast<float4*>(gS + s4) = *reinterpret_cast<const float4*>(gamma + s4);
        *reinterpret_cast<float4*>(bS + s4) = *reinterpret_cast<const float4*>(beta + s4);
    }
    __syncthreads();
    if (TID < 32) {
        float m = sums[TID] * (1.0f / 4096.0f);
        mean[TID] = m;
        rstd[TID] = rsqrtf(sqrs[TID] * (1.0f / 4096.0f) - m * m + 1e-5f);
    }
    __syncthreads();

    // Normalize + erf-GELU, vectorized: float4 LDS gamma/beta reads, float4 global stores.
    for (int r = 0; r < 32; ++r) {
        const float m = mean[r], rs = rstd[r];
        float* orow = out + ((size_t)(b * 512 + o0 + r)) * 4096;
        for (int s4 = TID * 4; s4 < 4096; s4 += 1024) {
            float4 g  = *reinterpret_cast<const float4*>(gS + s4);
            float4 be = *reinterpret_cast<const float4*>(bS + s4);
            float v0 = (bf2f(yT[(s4 + 0) * 34 + r]) - m) * rs * g.x + be.x;
            float v1 = (bf2f(yT[(s4 + 1) * 34 + r]) - m) * rs * g.y + be.y;
            float v2 = (bf2f(yT[(s4 + 2) * 34 + r]) - m) * rs * g.z + be.z;
            float v3 = (bf2f(yT[(s4 + 3) * 34 + r]) - m) * rs * g.w + be.w;
            float4 o;
            o.x = 0.5f * v0 * (1.0f + erff(v0 * 0.70710678118f));
            o.y = 0.5f * v1 * (1.0f + erff(v1 * 0.70710678118f));
            o.z = 0.5f * v2 * (1.0f + erff(v2 * 0.70710678118f));
            o.w = 0.5f * v3 * (1.0f + erff(v3 * 0.70710678118f));
            *reinterpret_cast<float4*>(orow + s4) = o;
        }
    }
}

extern "C" void kernel_launch(void* const* d_in, const int* in_sizes, int n_in,
                              void* d_out, int out_size, void* d_ws, size_t ws_size,
                              hipStream_t stream) {
    const float* x      = (const float*)d_in[0];   // [64,256,4096]
    const float* w1     = (const float*)d_in[1];   // [64,256]
    const float* w2     = (const float*)d_in[2];   // [4,64]
    const float* b2     = (const float*)d_in[3];   // [4]
    const float* conv_w = (const float*)d_in[4];   // [4,512,256]
    // d_in[5] = conv_b: mathematically a no-op (uniform shift along S cancels in LayerNorm)
    const float* gamma  = (const float*)d_in[6];   // [4096]
    const float* beta   = (const float*)d_in[7];   // [4096]
    float* out = (float*)d_out;                    // [64,512,4096] fp32

    char* ws = (char*)d_ws;
    unsigned short* Wb = (unsigned short*)ws;                        // 16 MiB bf16 [64][512][256]
    float* pooled = (float*)(ws + (size_t)16777216);                 // 64 KiB
    float* att    = (float*)(ws + (size_t)16777216 + 65536);         // 1 KiB

    dyn_pool<<<dim3(256, 64), 256, 0, stream>>>(x, pooled);
    dyn_attn<<<64, 256, 0, stream>>>(pooled, w1, w2, b2, att);
    dyn_mix<<<16384, 256, 0, stream>>>(att, conv_w, (unsigned*)Wb);

    const int shmem = 33792 + 278528 + 4 * 32 * (int)sizeof(float);  // 312832 B < 320 KB/WGP
    (void)hipFuncSetAttribute(reinterpret_cast<const void*>(dyn_main),
                              hipFuncAttributeMaxDynamicSharedMemorySize, shmem);
    dyn_main<<<dim3(16, 64), 256, shmem, stream>>>(x, Wb, gamma, beta, out);
}
